// Regressor_1168231104748
// MI455X (gfx1250) — matrix operand order repaired and checked by reading, not compiled, not run on hardware
//
#include <hip/hip_runtime.h>
#include <math.h>

// ---------------- problem constants (from setup_inputs) ----------------
#define N_NODES 50000
#define N_EDGES 800000
#define F_IN    301
#define K1PAD   320   // 301 padded to multiple of 32 for WMMA K

typedef __attribute__((ext_vector_type(16))) __bf16 v16bf;
typedef __attribute__((ext_vector_type(8)))  __bf16 v8bf;
typedef __attribute__((ext_vector_type(8)))  float  v8f;

typedef __attribute__((address_space(3))) const unsigned short lds_cus;

// ---------------- helpers ----------------
__device__ __forceinline__ unsigned short f2bf(float f) {
  unsigned u = __float_as_uint(f);
  u += 0x7FFFu + ((u >> 16) & 1u);            // round-to-nearest-even
  return (unsigned short)(u >> 16);
}
__device__ __forceinline__ float bf2f(unsigned short h) {
  return __uint_as_float(((unsigned)h) << 16);
}
__device__ __forceinline__ void atomicMaxF(float* addr, float val) {
  if (val >= 0.0f) atomicMax((int*)addr, __float_as_int(val));
  else             atomicMin((unsigned int*)addr, __float_as_uint(val));
}

__global__ void fill_f32(float* p, float v, size_t n) {
  size_t i = blockIdx.x * (size_t)blockDim.x + threadIdx.x;
  if (i < n) p[i] = v;
}

// ---------------- BatchNorm: per-column stats (training-mode batch stats) ----------------
__global__ void col_stats(const float* __restrict__ x, int n, int f,
                          float* __restrict__ mean, float* __restrict__ rstd) {
  int col = blockIdx.x;
  __shared__ float ssum[256], ssq[256];
  float s = 0.f, q = 0.f;
  for (int i = threadIdx.x; i < n; i += blockDim.x) {
    float v = x[(size_t)i * f + col];
    s += v; q += v * v;
  }
  ssum[threadIdx.x] = s; ssq[threadIdx.x] = q;
  __syncthreads();
  for (int off = blockDim.x >> 1; off > 0; off >>= 1) {
    if ((int)threadIdx.x < off) {
      ssum[threadIdx.x] += ssum[threadIdx.x + off];
      ssq[threadIdx.x]  += ssq[threadIdx.x + off];
    }
    __syncthreads();
  }
  if (threadIdx.x == 0) {
    float mu  = ssum[0] / n;
    float var = ssq[0] / n - mu * mu;
    mean[col] = mu;
    rstd[col] = rsqrtf(var + 1e-5f);
  }
}

// BN apply (+optional relu), cast to bf16, pad K to kpad with zeros
__global__ void bn_apply_cast_pad(const float* __restrict__ x,
                                  const float* __restrict__ mean, const float* __restrict__ rstd,
                                  const float* __restrict__ g, const float* __restrict__ b,
                                  unsigned short* __restrict__ out,
                                  int n, int f, int kpad, int do_relu) {
  size_t idx = blockIdx.x * (size_t)blockDim.x + threadIdx.x;
  size_t total = (size_t)n * kpad;
  if (idx >= total) return;
  int col = (int)(idx % kpad);
  int row = (int)(idx / kpad);
  float v = 0.f;
  if (col < f) {
    v = (x[(size_t)row * f + col] - mean[col]) * rstd[col] * g[col] + b[col];
    if (do_relu) v = fmaxf(v, 0.f);
  }
  out[idx] = f2bf(v);
}

// Build Bt[ntot][kpad] bf16 = concat(Wl, Wr) transposed; W shapes are [k][ncolsHalf]
__global__ void prep_bt(const float* __restrict__ Wl, const float* __restrict__ Wr,
                        int k, int kpad, int ncolsHalf, unsigned short* __restrict__ bt) {
  size_t idx = blockIdx.x * (size_t)blockDim.x + threadIdx.x;
  int ntot = 2 * ncolsHalf;
  size_t total = (size_t)ntot * kpad;
  if (idx >= total) return;
  int kk = (int)(idx % kpad);
  int nn = (int)(idx / kpad);
  float v = 0.f;
  if (kk < k)
    v = (nn < ncolsHalf) ? Wl[(size_t)kk * ncolsHalf + nn]
                         : Wr[(size_t)kk * ncolsHalf + (nn - ncolsHalf)];
  bt[idx] = f2bf(v);
}

// ---------------- WMMA GEMM with async-LDS-staged, double-buffered B ----------------
// Block = 8 waves = one 128x64 output tile. B tile (64 cols x 32 K, bf16 = 4 KB)
// is staged to LDS with global_load_async_to_lds_b128 (ASYNCcnt) and double-buffered
// so staging K-step k+1 overlaps WMMA of K-step k. A fragments stream from global,
// software-pipelined one K-step ahead.
// Fragment layouts used here (A/B in contiguous k order, not the ISA table order):
//   A 16x32 bf16: lane L holds row M=L%16; K=[h*16, h*16+16) contiguous, h=L/16
//   B 32x16 bf16: lane L holds col N=L%16; K=[h*16, h*16+16) contiguous
//   C 16x16 f32 : VGPR r -> row (h*8 + r), col L%16
__global__ void __launch_bounds__(256)
wmma_gemm(const unsigned short* __restrict__ A,
          const unsigned short* __restrict__ Bt,
          float* __restrict__ C,
          int M, int K, int Ncols) {
  __shared__ unsigned short Bsh[2][64 * 32];   // [buf][col*32 + k]  (8 KB)

  const int lane = threadIdx.x & 31;
  const int wave = threadIdx.x >> 5;
  const int nT = Ncols >> 6;                   // 64-wide column tiles
  const int mGroup = blockIdx.x / nT;
  const int nTile  = blockIdx.x - mGroup * nT;
  const int mTile  = mGroup * 8 + wave;
  const bool active = (mTile * 16 < M);
  const int half = lane >> 4;                  // 0 or 1
  const int l15  = lane & 15;

  // B staging assignment: 256 threads x 16B = 4 KB tile
  const int scol = threadIdx.x >> 2;           // 0..63
  const int sseg = threadIdx.x & 3;            // 0..3
  const unsigned short* bsrc = Bt + (size_t)(nTile * 64 + scol) * K + sseg * 8;
  const unsigned sOff = (unsigned)((scol * 32 + sseg * 8) * 2);  // bytes into tile
  const unsigned ldsB0 = (unsigned)(unsigned long long)(lds_cus*)&Bsh[0][0];
  const unsigned ldsB1 = (unsigned)(unsigned long long)(lds_cus*)&Bsh[1][0];

  const unsigned short* aRow = A + (size_t)(mTile * 16 + l15) * K;

  // stage K-step 0 into buffer 0
  {
    const unsigned short* sp = bsrc;
    unsigned dst = ldsB0 + sOff;
    asm volatile("global_load_async_to_lds_b128 %0, %1, off"
                 :: "v"(dst), "v"(sp) : "memory");
  }

  // prime A fragment pipeline
  v8bf aLo = {}, aHi = {};
  if (active) {
    aLo = *(const v8bf*)(aRow + half * 16);
    aHi = *(const v8bf*)(aRow + 8 + half * 16);
  }

  v8f c0 = {}, c1 = {}, c2 = {}, c3 = {};
  int p = 0;
  for (int k0 = 0; k0 < K; k0 += 32) {
    // all async writes into Bsh[p] landed; all reads of Bsh[p^1] from the
    // previous iteration completed before this barrier
    asm volatile("s_wait_asynccnt 0x0" ::: "memory");
    __syncthreads();

    // stage next K-step into the other buffer (overlaps compute below)
    if (k0 + 32 < K) {
      const unsigned short* sp = bsrc + (k0 + 32);
      unsigned dst = (p ? ldsB0 : ldsB1) + sOff;
      asm volatile("global_load_async_to_lds_b128 %0, %1, off"
                   :: "v"(dst), "v"(sp) : "memory");
    }

    // pack current A fragment, then issue next A loads early
    v16bf a;
    #pragma unroll
    for (int i = 0; i < 8; ++i) { a[i] = aLo[i]; a[8 + i] = aHi[i]; }
    if (active && (k0 + 32 < K)) {
      aLo = *(const v8bf*)(aRow + k0 + 32 + half * 16);
      aHi = *(const v8bf*)(aRow + k0 + 40 + half * 16);
    }

    if (active) {
      const unsigned short* bb = &Bsh[p][0];
      v16bf b0 = *(const v16bf*)(bb + (size_t)(( 0 + l15) * 32) + half * 16);
      v16bf b1 = *(const v16bf*)(bb + (size_t)((16 + l15) * 32) + half * 16);
      v16bf b2 = *(const v16bf*)(bb + (size_t)((32 + l15) * 32) + half * 16);
      v16bf b3 = *(const v16bf*)(bb + (size_t)((48 + l15) * 32) + half * 16);
      c0 = __builtin_amdgcn_wmma_f32_16x16x32_bf16(false, a, false, b0, (short)0, c0, false, false);
      c1 = __builtin_amdgcn_wmma_f32_16x16x32_bf16(false, a, false, b1, (short)0, c1, false, false);
      c2 = __builtin_amdgcn_wmma_f32_16x16x32_bf16(false, a, false, b2, (short)0, c2, false, false);
      c3 = __builtin_amdgcn_wmma_f32_16x16x32_bf16(false, a, false, b3, (short)0, c3, false, false);
    }
    p ^= 1;
  }

  if (active) {
    int rowBase = mTile * 16 + half * 8;
    int colBase = nTile * 64 + l15;
    #pragma unroll
    for (int r = 0; r < 8; ++r) {
      float* cp = C + (size_t)(rowBase + r) * Ncols + colBase;
      cp[0]  = c0[r];
      cp[16] = c1[r];
      cp[32] = c2[r];
      cp[48] = c3[r];
    }
  }
}

// ---------------- GATv2 edge phase ----------------
__global__ void calc_deg(const int* __restrict__ dst, float* __restrict__ deg, int E) {
  int e = blockIdx.x * blockDim.x + threadIdx.x;
  if (e < E) atomicAdd(deg + dst[e], 1.0f);
}

// logits[e,h] = sum_c leaky_relu(xl[src,h,c] + xr[dst,h,c] + ee[e,h,c], 0.2) * att[h,c]
// also segment-max into mbuf[dst,h]
__global__ void edge_logits(const int* __restrict__ src, const int* __restrict__ dst,
                            const float* __restrict__ ea,
                            const float* __restrict__ xl, int ldx,
                            const float* __restrict__ xr,
                            const float* __restrict__ We, const float* __restrict__ att,
                            float* __restrict__ elog, float* __restrict__ mbuf,
                            int E, int H, int Cc) {
  int idx = blockIdx.x * blockDim.x + threadIdx.x;
  if (idx >= E * H) return;
  int e = idx / H, h = idx - e * H;
  int s = src[e], d = dst[e];
  float a0 = ea[2 * e], a1 = ea[2 * e + 1];
  const float* pl = xl + (size_t)s * ldx + h * Cc;
  const float* pr = xr + (size_t)d * ldx + h * Cc;
  const float* w0 = We + h * Cc;
  const float* w1 = We + H * Cc + h * Cc;
  const float* at = att + h * Cc;
  float acc = 0.f;
  for (int c = 0; c < Cc; ++c) {
    float z = pl[c] + pr[c] + a0 * w0[c] + a1 * w1[c];
    z = (z > 0.f) ? z : 0.2f * z;
    acc += z * at[c];
  }
  elog[idx] = acc;
  atomicMaxF(mbuf + (size_t)d * H + h, acc);
}

// elog[e,h] <- exp(elog - m[dst]); segment-sum into sbuf
__global__ void edge_exp(const int* __restrict__ dst,
                         const float* __restrict__ mbuf, float* __restrict__ elog,
                         float* __restrict__ sbuf, int E, int H) {
  int idx = blockIdx.x * blockDim.x + threadIdx.x;
  if (idx >= E * H) return;
  int e = idx / H, h = idx - e * H;
  int d = dst[e];
  float v = expf(elog[idx] - mbuf[(size_t)d * H + h]);
  elog[idx] = v;
  atomicAdd(sbuf + (size_t)d * H + h, v);
}

// summed[dst,h,c] += (elog / (sbuf[dst,h]+1e-16)) * xl[src,h,c]
__global__ void edge_msg(const int* __restrict__ src, const int* __restrict__ dst,
                         const float* __restrict__ elog, const float* __restrict__ sbuf,
                         const float* __restrict__ xl, int ldx,
                         float* __restrict__ summed, int E, int H, int Cc) {
  size_t idx = blockIdx.x * (size_t)blockDim.x + threadIdx.x;
  size_t total = (size_t)E * H * Cc;
  if (idx >= total) return;
  int c = (int)(idx % Cc);
  size_t t = idx / Cc;
  int h = (int)(t % H);
  int e = (int)(t / H);
  int s = src[e], d = dst[e];
  float alpha = elog[(size_t)e * H + h] / (sbuf[(size_t)d * H + h] + 1e-16f);
  atomicAdd(summed + ((size_t)d * H + h) * Cc + c,
            alpha * xl[(size_t)s * ldx + h * Cc + c]);
}

// out[n,hc] = summed/max(deg,1) + bias[hc]   (in place on summed)
__global__ void finalize_mean_bias(float* __restrict__ summed, const float* __restrict__ deg,
                                   const float* __restrict__ bias, int n, int HC) {
  size_t idx = blockIdx.x * (size_t)blockDim.x + threadIdx.x;
  size_t total = (size_t)n * HC;
  if (idx >= total) return;
  int row = (int)(idx / HC);
  int hc  = (int)(idx % HC);
  summed[idx] = summed[idx] / fmaxf(deg[row], 1.0f) + bias[hc];
}

// layer-3 projections: xl3[n] = h3[n,:] . Wl3 ; xr3 likewise (k=128, scalar dot)
__global__ void proj3(const unsigned short* __restrict__ h3,
                      const float* __restrict__ Wl3, const float* __restrict__ Wr3,
                      float* __restrict__ xl3, float* __restrict__ xr3, int n, int k) {
  int i = blockIdx.x * blockDim.x + threadIdx.x;
  if (i >= n) return;
  const unsigned short* p = h3 + (size_t)i * k;
  float sl = 0.f, sr = 0.f;
  for (int c = 0; c < k; ++c) {
    float v = bf2f(p[c]);
    sl += v * Wl3[c];
    sr += v * Wr3[c];
  }
  xl3[i] = sl;
  xr3[i] = sr;
}

__global__ void final3(const float* __restrict__ summed, const float* __restrict__ deg,
                       const float* __restrict__ b3, float* __restrict__ y, int n) {
  int i = blockIdx.x * blockDim.x + threadIdx.x;
  if (i < n) y[i] = summed[i] / fmaxf(deg[i], 1.0f) + b3[0];
}

// ---------------- host orchestration ----------------
static inline unsigned nblk(size_t n, unsigned t) { return (unsigned)((n + t - 1) / t); }

extern "C" void kernel_launch(void* const* d_in, const int* in_sizes, int n_in,
                              void* d_out, int out_size, void* d_ws, size_t ws_size,
                              hipStream_t stream) {
  const float* x     = (const float*)d_in[0];
  const int*   ei    = (const int*)d_in[1];
  const float* ea    = (const float*)d_in[2];
  const float* bn0_g = (const float*)d_in[3];
  const float* bn0_b = (const float*)d_in[4];
  const float* Wl1   = (const float*)d_in[5];
  const float* Wr1   = (const float*)d_in[6];
  const float* We1   = (const float*)d_in[7];
  const float* att1  = (const float*)d_in[8];
  const float* b1    = (const float*)d_in[9];
  const float* bn1_g = (const float*)d_in[10];
  const float* bn1_b = (const float*)d_in[11];
  const float* Wl2   = (const float*)d_in[12];
  const float* Wr2   = (const float*)d_in[13];
  const float* We2   = (const float*)d_in[14];
  const float* att2  = (const float*)d_in[15];
  const float* b2    = (const float*)d_in[16];
  const float* bn2_g = (const float*)d_in[17];
  const float* bn2_b = (const float*)d_in[18];
  const float* Wl3   = (const float*)d_in[19];
  const float* Wr3   = (const float*)d_in[20];
  const float* We3   = (const float*)d_in[21];
  const float* att3  = (const float*)d_in[22];
  const float* b3    = (const float*)d_in[23];

  const int* src = ei;
  const int* dst = ei + N_EDGES;
  float* y = (float*)d_out;

  // bump allocator over workspace (~194 MB total)
  char* wp = (char*)d_ws;
  auto alloc = [&](size_t bytes) -> char* {
    char* p = wp;
    wp += (bytes + 255) & ~(size_t)255;
    return p;
  };
  unsigned short* Abuf  = (unsigned short*)alloc((size_t)N_NODES * K1PAD * 2); // bf16 activations
  unsigned short* Btbuf = (unsigned short*)alloc((size_t)512 * K1PAD * 2);     // bf16 weightsT
  float* G    = (float*)alloc((size_t)N_NODES * 512 * 4);   // GEMM out [xl|xr]
  float* SUM  = (float*)alloc((size_t)N_NODES * 256 * 4);   // segment sums / layer output
  float* ELOG = (float*)alloc((size_t)N_EDGES * 2 * 4);     // per-edge logits -> exp
  float* MBUF = (float*)alloc((size_t)N_NODES * 2 * 4);
  float* SBUF = (float*)alloc((size_t)N_NODES * 2 * 4);
  float* DEG  = (float*)alloc((size_t)N_NODES * 4);
  float* MEAN = (float*)alloc(512 * 4);
  float* RSTD = (float*)alloc(512 * 4);
  float* XL3  = (float*)alloc((size_t)N_NODES * 4);
  float* XR3  = (float*)alloc((size_t)N_NODES * 4);

  const unsigned T = 256;
  const float NEG_INF = -__builtin_huge_valf();
  const int mGroups = (N_NODES / 16 + 7) / 8;   // 391 groups of 128 rows

  // degrees (shared by all layers)
  fill_f32<<<nblk(N_NODES, T), T, 0, stream>>>(DEG, 0.f, N_NODES);
  calc_deg<<<nblk(N_EDGES, T), T, 0, stream>>>(dst, DEG, N_EDGES);

  // ---------- BN0 + cast ----------
  col_stats<<<F_IN, T, 0, stream>>>(x, N_NODES, F_IN, MEAN, RSTD);
  bn_apply_cast_pad<<<nblk((size_t)N_NODES * K1PAD, T), T, 0, stream>>>(
      x, MEAN, RSTD, bn0_g, bn0_b, Abuf, N_NODES, F_IN, K1PAD, 0);

  // ---------- layer 1: GEMM (K=320, Ncols=512 = [Wl1|Wr1]) ----------
  prep_bt<<<nblk((size_t)512 * K1PAD, T), T, 0, stream>>>(Wl1, Wr1, F_IN, K1PAD, 256, Btbuf);
  wmma_gemm<<<mGroups * (512 / 64), 256, 0, stream>>>(Abuf, Btbuf, G, N_NODES, K1PAD, 512);

  // edge phase: H=2, C=128; xl = G[:, :256] (ld 512), xr = G[:, 256:]
  fill_f32<<<nblk((size_t)N_NODES * 2, T), T, 0, stream>>>(MBUF, NEG_INF, (size_t)N_NODES * 2);
  fill_f32<<<nblk((size_t)N_NODES * 2, T), T, 0, stream>>>(SBUF, 0.f, (size_t)N_NODES * 2);
  fill_f32<<<nblk((size_t)N_NODES * 256, T), T, 0, stream>>>(SUM, 0.f, (size_t)N_NODES * 256);
  edge_logits<<<nblk((size_t)N_EDGES * 2, T), T, 0, stream>>>(
      src, dst, ea, G, 512, G + 256, We1, att1, ELOG, MBUF, N_EDGES, 2, 128);
  edge_exp<<<nblk((size_t)N_EDGES * 2, T), T, 0, stream>>>(dst, MBUF, ELOG, SBUF, N_EDGES, 2);
  edge_msg<<<nblk((size_t)N_EDGES * 2 * 128, T), T, 0, stream>>>(
      src, dst, ELOG, SBUF, G, 512, SUM, N_EDGES, 2, 128);
  finalize_mean_bias<<<nblk((size_t)N_NODES * 256, T), T, 0, stream>>>(SUM, DEG, b1, N_NODES, 256);

  // ---------- BN1 + relu + cast ----------
  col_stats<<<256, T, 0, stream>>>(SUM, N_NODES, 256, MEAN, RSTD);
  bn_apply_cast_pad<<<nblk((size_t)N_NODES * 256, T), T, 0, stream>>>(
      SUM, MEAN, RSTD, bn1_g, bn1_b, Abuf, N_NODES, 256, 256, 1);

  // ---------- layer 2: GEMM (K=256, Ncols=256 = [Wl2|Wr2]) ----------
  prep_bt<<<nblk((size_t)256 * 256, T), T, 0, stream>>>(Wl2, Wr2, 256, 256, 128, Btbuf);
  wmma_gemm<<<mGroups * (256 / 64), 256, 0, stream>>>(Abuf, Btbuf, G, N_NODES, 256, 256);

  // edge phase: H=1, C=128; xl = G[:, :128] (ld 256), xr = G[:, 128:]
  fill_f32<<<nblk(N_NODES, T), T, 0, stream>>>(MBUF, NEG_INF, N_NODES);
  fill_f32<<<nblk(N_NODES, T), T, 0, stream>>>(SBUF, 0.f, N_NODES);
  fill_f32<<<nblk((size_t)N_NODES * 128, T), T, 0, stream>>>(SUM, 0.f, (size_t)N_NODES * 128);
  edge_logits<<<nblk(N_EDGES, T), T, 0, stream>>>(
      src, dst, ea, G, 256, G + 128, We2, att2, ELOG, MBUF, N_EDGES, 1, 128);
  edge_exp<<<nblk(N_EDGES, T), T, 0, stream>>>(dst, MBUF, ELOG, SBUF, N_EDGES, 1);
  edge_msg<<<nblk((size_t)N_EDGES * 128, T), T, 0, stream>>>(
      src, dst, ELOG, SBUF, G, 256, SUM, N_EDGES, 1, 128);
  finalize_mean_bias<<<nblk((size_t)N_NODES * 128, T), T, 0, stream>>>(SUM, DEG, b2, N_NODES, 128);

  // ---------- BN2 + relu + cast ----------
  col_stats<<<128, T, 0, stream>>>(SUM, N_NODES, 128, MEAN, RSTD);
  bn_apply_cast_pad<<<nblk((size_t)N_NODES * 128, T), T, 0, stream>>>(
      SUM, MEAN, RSTD, bn2_g, bn2_b, Abuf, N_NODES, 128, 128, 1);

  // ---------- layer 3: scalar projections + edge phase (H=1, C=1) ----------
  proj3<<<nblk(N_NODES, T), T, 0, stream>>>(Abuf, Wl3, Wr3, XL3, XR3, N_NODES, 128);
  fill_f32<<<nblk(N_NODES, T), T, 0, stream>>>(MBUF, NEG_INF, N_NODES);
  fill_f32<<<nblk(N_NODES, T), T, 0, stream>>>(SBUF, 0.f, N_NODES);
  fill_f32<<<nblk(N_NODES, T), T, 0, stream>>>(SUM, 0.f, N_NODES);
  edge_logits<<<nblk(N_EDGES, T), T, 0, stream>>>(
      src, dst, ea, XL3, 1, XR3, We3, att3, ELOG, MBUF, N_EDGES, 1, 1);
  edge_exp<<<nblk(N_EDGES, T), T, 0, stream>>>(dst, MBUF, ELOG, SBUF, N_EDGES, 1);
  edge_msg<<<nblk(N_EDGES, T), T, 0, stream>>>(
      src, dst, ELOG, SBUF, XL3, 1, SUM, N_EDGES, 1, 1);
  final3<<<nblk(N_NODES, T), T, 0, stream>>>(SUM, DEG, b3, y, N_NODES);
}
